// StrataLayer_59021440581647
// MI455X (gfx1250) — matrix operand order repaired and checked by reading, not compiled, not run on hardware
//
#include <hip/hip_runtime.h>
#include <math.h>

// ---------------------------------------------------------------------------
// CDNA5 (gfx1250) Strata forward: 3x 6-layer transformer encoder (bf16 WMMA)
// + sinusoidal PE prologue + hierarchy-softmax combine epilogue.
// ---------------------------------------------------------------------------

typedef __attribute__((ext_vector_type(16))) __bf16 v16bf;
typedef __attribute__((ext_vector_type(2)))  __bf16 bf16x2;
typedef __attribute__((ext_vector_type(8)))  float  v8f;

#define S_IN   32
#define BATCH  4
#define DMODEL 512
#define FFDIM  1024
#define NHEADS 8
#define HD     64
#define SEQ    32       // encoder sequence length == block_size
#define NBATCH 128      // encoder batch = S_IN * BATCH
#define MROWS  4096     // SEQ * NBATCH
#define NLAYER 6
#define LOUT   256

// native packed f32 -> bf16 (RTNE); lowers to v_cvt_pk_bf16_f32 on gfx1250
__device__ __forceinline__ bf16x2 pk_bf16(float x, float y) {
  bf16x2 r;
  r.x = (__bf16)x;
  r.y = (__bf16)y;
  return r;
}

union Frag16 {                    // 32-byte A/B fragment (16 x bf16)
  v16bf v;
  unsigned int d[8];
  uint4 q[2];
};

// ---------------------------------------------------------------------------
// Generic WMMA GEMM:  C[M,N] = act( A[M,K](f32) * W[N,K](f32)^T + bias[N] )
// Block: 128 threads (4 waves), tile 64x64, wave tile 32x32 (2x2 WMMA 16x16x32)
// ---------------------------------------------------------------------------
#define BM 64
#define BN 64
#define BK 32
#define LDST (BK + 8)             // padded ushort row stride (80B, 16B aligned)

__global__ __launch_bounds__(128)
void gemm_bf16_wmma(const float* __restrict__ A,
                    const float* __restrict__ W,
                    const float* __restrict__ bias,
                    float* __restrict__ C,
                    int M, int N, int K, int relu)
{
  __shared__ unsigned short As[BM * LDST];
  __shared__ unsigned short Bs[BN * LDST];

  const int tid  = threadIdx.x;
  const int lane = tid & 31;
  const int wave = tid >> 5;
  const int bm = blockIdx.y * BM;
  const int bn = blockIdx.x * BN;
  const int wm = (wave >> 1) * 32;          // wave M offset inside tile
  const int wn = (wave & 1)  * 32;          // wave N offset inside tile

  v8f acc[2][2] = {};

  const int lrow  = tid >> 1;               // 0..63
  const int lhalf = (tid & 1) * 16;         // 0 or 16 (K columns)
  const int h   = lane >> 4;                // lane half (0/1)
  const int r16 = lane & 15;

  for (int k0 = 0; k0 < K; k0 += BK) {
    // ---- stage A tile (f32 -> bf16 via v_cvt_pk_bf16_f32) ----
    const float* ga = A + (size_t)(bm + lrow) * K + k0 + lhalf;
    unsigned short* sa = As + lrow * LDST + lhalf;
    #pragma unroll
    for (int j = 0; j < 16; j += 4) {
      float4 f = *(const float4*)(ga + j);
      *(bf16x2*)(sa + j)     = pk_bf16(f.x, f.y);
      *(bf16x2*)(sa + j + 2) = pk_bf16(f.z, f.w);
    }
    // ---- stage B tile (weights [N,K] row-major, f32 -> bf16) ----
    const float* gb = W + (size_t)(bn + lrow) * K + k0 + lhalf;
    unsigned short* sb = Bs + lrow * LDST + lhalf;
    #pragma unroll
    for (int j = 0; j < 16; j += 4) {
      float4 f = *(const float4*)(gb + j);
      *(bf16x2*)(sb + j)     = pk_bf16(f.x, f.y);
      *(bf16x2*)(sb + j + 2) = pk_bf16(f.z, f.w);
    }
    __syncthreads();

    if (k0 + BK < K) {                       // prefetch next K tile into L2/L0
      __builtin_prefetch(ga + BK, 0, 1);
      __builtin_prefetch(gb + BK, 0, 1);
    }

    // ---- build fragments, contiguous K = 16*h + s for both A and B ----
    Frag16 a[2], b[2];
    #pragma unroll
    for (int i = 0; i < 2; ++i) {
      // A 16x32 bf16: lane half h, row = r16; elems = 16 contiguous K at 16*h
      const unsigned int* rowp =
          (const unsigned int*)(As + (wm + 16 * i + r16) * LDST);
      #pragma unroll
      for (int p = 0; p < 8; ++p)
        a[i].d[p] = rowp[8 * h + p];
    }
    #pragma unroll
    for (int j = 0; j < 2; ++j) {
      // B 32x16 bf16: lane = column N, elems = 16 contiguous K at 16*h
      const uint4* rp =
          (const uint4*)(Bs + (wn + 16 * j + r16) * LDST + 16 * h);
      b[j].q[0] = rp[0];
      b[j].q[1] = rp[1];
    }

    #pragma unroll
    for (int i = 0; i < 2; ++i)
      #pragma unroll
      for (int j = 0; j < 2; ++j)
        acc[i][j] = __builtin_amdgcn_wmma_f32_16x16x32_bf16(
            false, a[i].v, false, b[j].v, (short)0, acc[i][j], false, false);

    __syncthreads();
  }

  // ---- epilogue: bias, optional relu, store f32 ----
  #pragma unroll
  for (int i = 0; i < 2; ++i)
    #pragma unroll
    for (int j = 0; j < 2; ++j)
      #pragma unroll
      for (int r = 0; r < 8; ++r) {
        int row = bm + wm + 16 * i + 8 * h + r;
        int col = bn + wn + 16 * j + r16;
        float v = acc[i][j][r] + bias[col];
        if (relu) v = fmaxf(v, 0.0f);
        C[(size_t)row * N + col] = v;
      }
}

// ---------------------------------------------------------------------------
// prep: replicate python double-precision position accumulation
// ---------------------------------------------------------------------------
__global__ void prep_idx(const int* lenPtr, const int* bsPtr,
                         int* mArr, int* lowArr, int* startArr, int* highArr)
{
  if (threadIdx.x == 0 && blockIdx.x == 0) {
    int L = lenPtr[0];
    int half = bsPtr[0] / 2;
    double step = (S_IN > 1) ? (double)(L - 1) / (double)(S_IN - 1) : 0.0;
    double pos = 0.0;
    for (int i = 0; i < S_IN; ++i) {
      int mi = (int)pos;                       // trunc == floor (pos >= 0)
      mArr[i] = mi;
      lowArr[i]  = mi - half;
      highArr[i] = mi + half;
      double st = ceil(pos - (double)half);
      startArr[i] = (st < 0.0) ? 0 : (int)st;
      pos += step;
    }
  }
}

// ---------------------------------------------------------------------------
// PE + flatten: flat[t, s*B+b, :] = inputs[s,b,:] + pe(low[s]+t, :)
// ---------------------------------------------------------------------------
__global__ __launch_bounds__(128)
void pe_encode(const float* __restrict__ inputs, const int* __restrict__ lowArr,
               float* __restrict__ flatF)
{
  int m  = blockIdx.x;                 // 0..4095 = t*128 + s*4 + b
  int t  = m >> 7;
  int nb = m & 127;
  int s  = nb >> 2;
  int b  = nb & 3;
  float pos = (float)(lowArr[s] + t);
  const float c = logf(10000.0f) / (float)DMODEL;
  for (int dd = threadIdx.x; dd < DMODEL; dd += 128) {
    int j2 = dd & ~1;
    float ang = pos * expf(-(float)j2 * c);
    float pe = (dd & 1) ? cosf(ang) : sinf(ang);
    flatF[(size_t)m * DMODEL + dd] =
        inputs[(size_t)((s << 2) + b) * DMODEL + dd] + pe;
  }
}

// ---------------------------------------------------------------------------
// Attention: one wave per (batch nb, head h). 32x32 scores, softmax, AV.
// ---------------------------------------------------------------------------
__global__ __launch_bounds__(32)
void attention(const float* __restrict__ qkv, float* __restrict__ attnO)
{
  __shared__ float qs[SEQ][HD], ks[SEQ][HD], vs[SEQ][HD];
  int nb = blockIdx.x >> 3;
  int hh = blockIdx.x & 7;
  int lane = threadIdx.x;

  const float* base = qkv + (size_t)(lane * NBATCH + nb) * (3 * DMODEL) + hh * HD;
  #pragma unroll
  for (int j = 0; j < HD; j += 4) {
    *(float4*)&qs[lane][j] = *(const float4*)(base + j);
    *(float4*)&ks[lane][j] = *(const float4*)(base + DMODEL + j);
    *(float4*)&vs[lane][j] = *(const float4*)(base + 2 * DMODEL + j);
  }
  __syncthreads();

  float sc[SEQ];
  float mx = -3.0e38f;
  for (int t = 0; t < SEQ; ++t) {
    float d = 0.0f;
    #pragma unroll
    for (int j = 0; j < HD; ++j) d += qs[lane][j] * ks[t][j];
    d *= 0.125f;                      // 1/sqrt(64)
    sc[t] = d;
    mx = fmaxf(mx, d);
  }
  float sum = 0.0f;
  for (int t = 0; t < SEQ; ++t) { sc[t] = expf(sc[t] - mx); sum += sc[t]; }
  float inv = 1.0f / sum;

  float o[HD];
  #pragma unroll
  for (int j = 0; j < HD; ++j) o[j] = 0.0f;
  for (int t = 0; t < SEQ; ++t) {
    float p = sc[t] * inv;
    #pragma unroll
    for (int j = 0; j < HD; ++j) o[j] += p * vs[t][j];
  }
  float* out = attnO + (size_t)(lane * NBATCH + nb) * DMODEL + hh * HD;
  #pragma unroll
  for (int j = 0; j < HD; j += 4)
    *(float4*)(out + j) = make_float4(o[j], o[j + 1], o[j + 2], o[j + 3]);
}

// ---------------------------------------------------------------------------
// out = LayerNorm(x + a) * g + beta    (one wave32 per 512-wide row)
// ---------------------------------------------------------------------------
__global__ __launch_bounds__(128)
void add_layernorm(const float* x, const float* a,
                   const float* g, const float* beta, float* out)
{
  int row  = blockIdx.x * 4 + (threadIdx.x >> 5);
  int lane = threadIdx.x & 31;
  const float* xr = x + (size_t)row * DMODEL;
  const float* ar = a + (size_t)row * DMODEL;

  float vals[16];
  float s = 0.0f;
  #pragma unroll
  for (int k = 0; k < 16; ++k) {
    float v = xr[lane + 32 * k] + ar[lane + 32 * k];
    vals[k] = v; s += v;
  }
  #pragma unroll
  for (int off = 16; off > 0; off >>= 1) s += __shfl_xor(s, off, 32);
  float mu = s * (1.0f / DMODEL);

  float vv = 0.0f;
  #pragma unroll
  for (int k = 0; k < 16; ++k) { float d = vals[k] - mu; vv += d * d; }
  #pragma unroll
  for (int off = 16; off > 0; off >>= 1) vv += __shfl_xor(vv, off, 32);
  float inv = rsqrtf(vv * (1.0f / DMODEL) + 1e-5f);

  #pragma unroll
  for (int k = 0; k < 16; ++k) {
    int c = lane + 32 * k;
    out[(size_t)row * DMODEL + c] = (vals[k] - mu) * inv * g[c] + beta[c];
  }
}

// ---------------------------------------------------------------------------
// Hierarchy combine: masked softmax over S_IN windows + mod-32 gather.
// ---------------------------------------------------------------------------
__global__ __launch_bounds__(256)
void combine(const float* __restrict__ vert, const float* __restrict__ horz,
             const float* __restrict__ o0, const float* __restrict__ o1,
             const float* __restrict__ o2,
             const int* __restrict__ mArr, const int* __restrict__ startArr,
             const int* __restrict__ highArr, float* __restrict__ out)
{
  int t = blockIdx.x >> 2;
  int b = blockIdx.x & 3;
  for (int dd = threadIdx.x; dd < DMODEL; dd += 256) {
    float lg[S_IN];
    bool  ok[S_IN];
    float mx = -3.0e38f;
    bool any = false;
    #pragma unroll
    for (int s = 0; s < S_IN; ++s) {
      ok[s] = (startArr[s] <= t) && (t <= highArr[s]);
      float df = (float)(t - mArr[s]);
      size_t vi = (size_t)((s << 2) + b) * DMODEL + dd;
      float v = vert[vi], hz = horz[vi];
      float sv = v * v * expf(-(df * df) / (hz * hz + 1e-10f));
      lg[s] = sv;
      if (ok[s]) { mx = fmaxf(mx, sv); any = true; }
    }
    if (!any) mx = 0.0f;
    float den = 0.0f;
    float ee[S_IN];
    #pragma unroll
    for (int s = 0; s < S_IN; ++s) {
      ee[s] = ok[s] ? expf(lg[s] - mx) : 0.0f;
      den += ee[s];
    }
    float inv = (den > 0.0f) ? (1.0f / den) : 0.0f;
    float r0 = 0.0f, r1 = 0.0f, r2 = 0.0f;
    #pragma unroll
    for (int s = 0; s < S_IN; ++s) {
      float scl = ee[s] * inv;
      int idx = (t - highArr[s]) & 31;       // mod 32 (two's complement safe)
      size_t off = (size_t)(idx * NBATCH + (s << 2) + b) * DMODEL + dd;
      r0 += scl * o0[off];
      r1 += scl * o1[off];
      r2 += scl * o2[off];
    }
    size_t oo = (size_t)(t * BATCH + b) * DMODEL + dd;
    out[oo]                         = r0;
    out[(size_t)LOUT * BATCH * DMODEL + oo]     = r1;
    out[(size_t)2 * LOUT * BATCH * DMODEL + oo] = r2;
  }
}

// ---------------------------------------------------------------------------
// Host orchestration
// ---------------------------------------------------------------------------
extern "C" void kernel_launch(void* const* d_in, const int* in_sizes, int n_in,
                              void* d_out, int out_size, void* d_ws, size_t ws_size,
                              hipStream_t stream)
{
  const float* inputs     = (const float*)d_in[0];
  const float* vertical   = (const float*)d_in[1];
  const float* horizontal = (const float*)d_in[2];
  const int*   lenPtr     = (const int*)d_in[39];
  const int*   bsPtr      = (const int*)d_in[40];

  // params_out / params_v / params_h, 12 tensors each in make_params order:
  // Wqkv, bqkv, Wo, bo, W1, b1, W2, b2, g1, bln1, g2, bln2
  const float* P[3][12];
  for (int e = 0; e < 3; ++e)
    for (int k = 0; k < 12; ++k)
      P[e][k] = (const float*)d_in[3 + e * 12 + k];

  // ---- workspace layout ----
  char* w = (char*)d_ws;
  int* mArr     = (int*)w;
  int* lowArr   = mArr + 32;
  int* startArr = mArr + 64;
  int* highArr  = mArr + 96;
  const size_t szAct = (size_t)MROWS * DMODEL;              // 2,097,152 f32
  float* flatF = (float*)(w + 1024);
  float* curF  = flatF + szAct;
  float* qkvF  = curF  + szAct;                             // M x 1536
  float* attnF = qkvF  + (size_t)MROWS * 3 * DMODEL;
  float* resF  = attnF + szAct;
  float* ffF   = resF  + szAct;                             // M x 1024
  float* outE[3];
  outE[0] = ffF + (size_t)MROWS * FFDIM;
  outE[1] = outE[0] + szAct;
  outE[2] = outE[1] + szAct;

  prep_idx<<<1, 32, 0, stream>>>(lenPtr, bsPtr, mArr, lowArr, startArr, highArr);
  pe_encode<<<MROWS, 128, 0, stream>>>(inputs, lowArr, flatF);

  dim3 blk(128);
  dim3 gQKV(3 * DMODEL / BN, MROWS / BM);   // 24 x 64
  dim3 gD  (DMODEL / BN,     MROWS / BM);   //  8 x 64
  dim3 gFF (FFDIM / BN,      MROWS / BM);   // 16 x 64

  for (int e = 0; e < 3; ++e) {
    hipMemcpyAsync(curF, flatF, szAct * sizeof(float),
                   hipMemcpyDeviceToDevice, stream);
    const float* Wqkv = P[e][0]; const float* bqkv = P[e][1];
    const float* Wo   = P[e][2]; const float* bo   = P[e][3];
    const float* W1   = P[e][4]; const float* b1   = P[e][5];
    const float* W2   = P[e][6]; const float* b2   = P[e][7];
    const float* g1   = P[e][8]; const float* bln1 = P[e][9];
    const float* g2   = P[e][10]; const float* bln2 = P[e][11];

    for (int l = 0; l < NLAYER; ++l) {
      // QKV projection
      gemm_bf16_wmma<<<gQKV, blk, 0, stream>>>(
          curF, Wqkv + (size_t)l * 3 * DMODEL * DMODEL, bqkv + l * 3 * DMODEL,
          qkvF, MROWS, 3 * DMODEL, DMODEL, 0);
      // Attention
      attention<<<NBATCH * NHEADS, 32, 0, stream>>>(qkvF, attnF);
      // Output projection
      gemm_bf16_wmma<<<gD, blk, 0, stream>>>(
          attnF, Wo + (size_t)l * DMODEL * DMODEL, bo + l * DMODEL,
          resF, MROWS, DMODEL, DMODEL, 0);
      // x = LN(x + attn)
      add_layernorm<<<MROWS / 4, 128, 0, stream>>>(
          curF, resF, g1 + l * DMODEL, bln1 + l * DMODEL, curF);
      // FF up + relu
      gemm_bf16_wmma<<<gFF, blk, 0, stream>>>(
          curF, W1 + (size_t)l * FFDIM * DMODEL, b1 + l * FFDIM,
          ffF, MROWS, FFDIM, DMODEL, 1);
      // FF down
      gemm_bf16_wmma<<<gD, blk, 0, stream>>>(
          ffF, W2 + (size_t)l * DMODEL * FFDIM, b2 + l * DMODEL,
          resF, MROWS, DMODEL, FFDIM, 0);
      // x = LN(x + ff)
      float* dst = (l == NLAYER - 1) ? outE[e] : curF;
      add_layernorm<<<MROWS / 4, 128, 0, stream>>>(
          curF, resF, g2 + l * DMODEL, bln2 + l * DMODEL, dst);
    }
  }

  combine<<<LOUT * BATCH, 256, 0, stream>>>(
      vertical, horizontal, outE[0], outE[1], outE[2],
      mArr, startArr, highArr, (float*)d_out);
}
